// CenterNCALoss_76639396429875
// MI455X (gfx1250) — matrix operand order repaired and checked by reading, not compiled, not run on hardware
//
#include <hip/hip_runtime.h>
#include <hip/hip_bf16.h>
#include <math.h>
#include <stdint.h>

typedef float  v4f  __attribute__((ext_vector_type(4)));
typedef float  v8f  __attribute__((ext_vector_type(8)));
typedef __bf16 v16bf __attribute__((ext_vector_type(16)));

#define ALPHA_F 16.0f
#define NUM_CLASS 1024
#define DIMS 128

// ---------------- workspace layout ----------------
// floats:
//   [0 .. 131071]      f32 class sums (scatter target)
//   [131072 .. 132095] counts
//   [132096 .. 133119] yy = ||center||^2 (exact f32)
// bytes:
//   532480  : CH bf16[1024*128] (centers hi)   262144 B  (32B aligned)
//   794624  : CL bf16[1024*128] (centers lo)   262144 B  (32B aligned)
//   1056768 : double accum[3] = {sum_loss, sum_pos, sum_dist}
#define WS_CSUM_F   0
#define WS_COUNTS_F 131072
#define WS_YY_F     132096
#define WS_CH_B     532480
#define WS_CL_B     794624
#define WS_ACC_B    1056768
#define WS_ZERO_F   132096

// LDS layout inside main kernel (bytes)
#define LDS_CH0 0
#define LDS_CL0 4096
#define LDS_CH1 8192
#define LDS_CL1 12288
#define LDS_YY  16384       // 1024 floats
#define LDS_TOTAL (16384 + 4096)

__global__ void nca_zero_kernel(float* p, int ntot, double* acc) {
  int i = blockIdx.x * blockDim.x + threadIdx.x;
  if (i < ntot) p[i] = 0.0f;
  if (i < 3) acc[i] = 0.0;
}

// one thread per (row, 4-dim chunk): atomic scatter-add into class sums
__global__ void nca_scatter_kernel(const float* __restrict__ X,
                                   const int* __restrict__ T,
                                   float* __restrict__ csum,
                                   float* __restrict__ counts, int n) {
  int idx = blockIdx.x * blockDim.x + threadIdx.x;
  if (idx >= n * (DIMS / 4)) return;
  int row = idx >> 5;
  int chunk = idx & 31;
  int t = T[row];
  const float* src = X + (size_t)row * DIMS + chunk * 4;
  float* dst = csum + (size_t)t * DIMS + chunk * 4;
  atomicAdd(dst + 0, src[0]);
  atomicAdd(dst + 1, src[1]);
  atomicAdd(dst + 2, src[2]);
  atomicAdd(dst + 3, src[3]);
  if (chunk == 0) atomicAdd(counts + t, 1.0f);
}

// centers = sums/counts; split into bf16 hi/lo planes; yy[c] = ||center||^2 (f32)
__global__ void nca_finalize_centers(const float* __restrict__ csum,
                                     const float* __restrict__ counts,
                                     float* __restrict__ yy,
                                     __bf16* __restrict__ CH,
                                     __bf16* __restrict__ CL) {
  int c = blockIdx.x;
  int d2 = threadIdx.x;                 // 64 threads, 2 dims each
  float cnt = counts[c];
  float v0 = csum[c * DIMS + 2 * d2]     / cnt;
  float v1 = csum[c * DIMS + 2 * d2 + 1] / cnt;
  __bf16 h0 = (__bf16)v0, h1 = (__bf16)v1;
  CH[c * DIMS + 2 * d2]     = h0;
  CH[c * DIMS + 2 * d2 + 1] = h1;
  CL[c * DIMS + 2 * d2]     = (__bf16)(v0 - (float)h0);
  CL[c * DIMS + 2 * d2 + 1] = (__bf16)(v1 - (float)h1);
  float sq = v0 * v0 + v1 * v1;
  #pragma unroll
  for (int m = 16; m >= 1; m >>= 1) sq += __shfl_xor(sq, m, 32);
  __shared__ float partial[2];
  int lane = threadIdx.x & 31, w = threadIdx.x >> 5;
  if (lane == 0) partial[w] = sq;
  __syncthreads();
  if (threadIdx.x == 0) yy[c] = partial[0] + partial[1];
}

// fused distance pass: split-bf16 WMMA (3 terms) + online logsumexp epilogue.
// Centers tiles double-buffered in LDS via async global->LDS copies (ASYNCcnt);
// B fragments software-pipelined one K-block ahead out of LDS.
// 8 waves/block, one 16-row tile per wave, sweeping all 64 center col-tiles.
__global__ __launch_bounds__(256) void nca_main_kernel(
    const float* __restrict__ X, const int* __restrict__ T,
    const __bf16* __restrict__ CH, const __bf16* __restrict__ CL,
    const float* __restrict__ YY, double* __restrict__ accum) {
  __shared__ __align__(64) unsigned char smem[LDS_TOTAL];

  const int tid   = threadIdx.x;
  const int lane  = tid & 31;
  const int wave  = tid >> 5;
  const int tile  = blockIdx.x * 8 + wave;
  const int rbase = tile * 16;
  const int half  = lane >> 4;     // 0: lanes 0-15, 1: lanes 16-31
  const int l15   = lane & 15;

  // --- async copy of one 16-center tile (CH 4KB + CL 4KB) into an LDS buffer ---
  auto issue_tile = [&](int ct, int buf) {
    unsigned lds_ch = (unsigned)(uintptr_t)(smem + (buf ? LDS_CH1 : LDS_CH0)) + tid * 16;
    unsigned lds_cl = (unsigned)(uintptr_t)(smem + (buf ? LDS_CL1 : LDS_CL0)) + tid * 16;
    unsigned long long g_ch =
        (unsigned long long)(uintptr_t)((const unsigned char*)CH + (size_t)ct * 4096 + tid * 16);
    unsigned long long g_cl =
        (unsigned long long)(uintptr_t)((const unsigned char*)CL + (size_t)ct * 4096 + tid * 16);
    asm volatile("global_load_async_to_lds_b128 %0, %1, off"
                 :: "v"(lds_ch), "v"(g_ch) : "memory");
    asm volatile("global_load_async_to_lds_b128 %0, %1, off"
                 :: "v"(lds_cl), "v"(g_cl) : "memory");
  };

  // prologue: kick off tile 0 and stage yy into LDS
  issue_tile(0, 0);
  float* syy = (float*)(smem + LDS_YY);
  #pragma unroll
  for (int i = 0; i < NUM_CLASS / 256; ++i) syy[tid + i * 256] = YY[tid + i * 256];

  // ---- A fragments: bf16 16x32 layout, 4 K-blocks of 32 ----
  // lane (half,l15): M=l15; VGPR0-7 hold K = kb*32 + 16*half + {0..15}
  v16bf ah[4], al[4];
  float xxp = 0.0f;
  const float* xrow = X + (size_t)(rbase + l15) * DIMS;
  #pragma unroll
  for (int kb = 0; kb < 4; ++kb) {
    const int b0 = kb * 32 + 16 * half;
    float f[16];
    *(v4f*)(f + 0)  = *(const v4f*)(xrow + b0);
    *(v4f*)(f + 4)  = *(const v4f*)(xrow + b0 + 4);
    *(v4f*)(f + 8)  = *(const v4f*)(xrow + b0 + 8);
    *(v4f*)(f + 12) = *(const v4f*)(xrow + b0 + 12);
    #pragma unroll
    for (int j = 0; j < 16; ++j) {
      float x = f[j];
      xxp += x * x;                       // exact f32 row norm contribution
      __bf16 h = (__bf16)x;
      ah[kb][j] = h;
      al[kb][j] = (__bf16)(x - (float)h); // residual, exact
    }
  }
  xxp += __shfl_xor(xxp, 16, 32);         // lanes r, r+16 hold xx[rbase + (lane&15)]

  float xxv[8];
  int tgt[8];
  #pragma unroll
  for (int v = 0; v < 8; ++v) {
    xxv[v] = __shfl(xxp, v + half * 8, 32);
    tgt[v] = T[rbase + v + half * 8];
  }

  float runM[8], runS[8], sdist[8], pos[8];
  #pragma unroll
  for (int v = 0; v < 8; ++v) {
    runM[v] = -INFINITY; runS[v] = 0.0f; sdist[v] = 0.0f; pos[v] = 0.0f;
  }

  // ---- sweep 64 column tiles of 16 centers, LDS double-buffered ----
  for (int ct = 0; ct < NUM_CLASS / 16; ++ct) {
    // tile ct's async writes (issued last iteration) complete, then block-sync.
    asm volatile("s_wait_asynccnt 0x0" ::: "memory");
    __syncthreads();
    // prefetch tile ct+1 into the other buffer; safe: every wave has passed the
    // barrier, so all reads of that buffer's previous contents are done.
    if (ct + 1 < NUM_CLASS / 16) issue_tile(ct + 1, (ct + 1) & 1);

    const unsigned char* chb = smem + ((ct & 1) ? LDS_CH1 : LDS_CH0);
    const unsigned char* clb = smem + ((ct & 1) ? LDS_CL1 : LDS_CL0);
    // B fragment (bf16 32x16): lane (half,l15): N=l15,
    // VGPR j holds K = 16*half + {2j, 2j+1} -> 16 contiguous bf16 per lane
    const __bf16* chrow = (const __bf16*)(chb + l15 * 256 + half * 32);
    const __bf16* clrow = (const __bf16*)(clb + l15 * 256 + half * 32);
    const float yyl = syy[ct * 16 + l15];
    const int col = ct * 16 + l15;

    // depth-2 software pipeline of B fragments out of LDS: each load pair is
    // issued ~3 WMMAs before its use, so dscnt waits are (mostly) satisfied.
    v16bf bh = *(const v16bf*)(chrow);
    v16bf bl = *(const v16bf*)(clrow);
    v8f acc = {0.f, 0.f, 0.f, 0.f, 0.f, 0.f, 0.f, 0.f};
    #pragma unroll
    for (int kb = 0; kb < 4; ++kb) {
      v16bf nbh, nbl;
      if (kb < 3) {
        nbh = *(const v16bf*)(chrow + (kb + 1) * 32);
        nbl = *(const v16bf*)(clrow + (kb + 1) * 32);
      }
      // x.c ~= xh.ch + xl.ch + xh.cl  (xl.cl ~ 2^-16, below split resolution)
      acc = __builtin_amdgcn_wmma_f32_16x16x32_bf16(
          false, ah[kb], false, bh, (short)0, acc, false, false);
      acc = __builtin_amdgcn_wmma_f32_16x16x32_bf16(
          false, al[kb], false, bh, (short)0, acc, false, false);
      acc = __builtin_amdgcn_wmma_f32_16x16x32_bf16(
          false, ah[kb], false, bl, (short)0, acc, false, false);
      if (kb < 3) { bh = nbh; bl = nbl; }   // renamed away by full unroll
    }

    // epilogue: element (v,lane) = dist(row rbase+v+8*half, col ct*16+l15)
    #pragma unroll
    for (int v = 0; v < 8; ++v) {
      float d = fmaf(-2.0f, acc[v], xxv[v] + yyl);
      sdist[v] += d;
      pos[v] += (col == tgt[v]) ? d : 0.0f;
      float t = -ALPHA_F * d;
      // single-exp online logsumexp update
      float delta = t - runM[v];
      float e = __expf(-fabsf(delta));       // exp(-|t-M|)
      runS[v] = (delta > 0.0f) ? fmaf(runS[v], e, 1.0f) : (runS[v] + e);
      runM[v] = fmaxf(runM[v], t);
    }
  }

  // ---- reduce across the 16 lanes sharing each row; f64 global atomics ----
  #pragma unroll
  for (int v = 0; v < 8; ++v) {
    #pragma unroll
    for (int m = 1; m <= 8; m <<= 1) {
      float M2 = __shfl_xor(runM[v], m, 32);
      float S2 = __shfl_xor(runS[v], m, 32);
      float nm = fmaxf(runM[v], M2);
      runS[v] = runS[v] * __expf(runM[v] - nm) + S2 * __expf(M2 - nm);
      runM[v] = nm;
      sdist[v] += __shfl_xor(sdist[v], m, 32);
      pos[v]   += __shfl_xor(pos[v], m, 32);
    }
    if (l15 == 0) {
      float lse = runM[v] + __logf(runS[v]);     // log sum_j exp(-alpha*dist)
      float loss_i = ALPHA_F * pos[v] + lse;     // global base cancels analytically
      atomicAdd(&accum[0], (double)loss_i);
      atomicAdd(&accum[1], (double)pos[v]);
      atomicAdd(&accum[2], (double)sdist[v]);
    }
  }
}

__global__ void nca_writeout(const double* __restrict__ accum,
                             float* __restrict__ out, int n) {
  if (threadIdx.x != 0 || blockIdx.x != 0) return;
  double dn = (double)n;
  double ni = dn / (double)NUM_CLASS;
  double sum_loss = accum[0], sum_pos = accum[1], sum_dist = accum[2];
  out[0] = (float)(sum_loss / dn);                                  // loss
  out[1] = 0.0f;                                                    // prec
  out[2] = (float)((ni / (ni - 1.0)) * (sum_pos / dn));             // dist_ap
  out[3] = (float)((sum_dist - sum_pos) / (dn * (NUM_CLASS - 1)));  // dist_an
}

extern "C" void kernel_launch(void* const* d_in, const int* in_sizes, int n_in,
                              void* d_out, int out_size, void* d_ws, size_t ws_size,
                              hipStream_t stream) {
  const float* X = (const float*)d_in[0];
  const int*   T = (const int*)d_in[1];
  const int n = in_sizes[0] / DIMS;                 // 65536

  float*  ws     = (float*)d_ws;
  float*  csum   = ws + WS_CSUM_F;
  float*  counts = ws + WS_COUNTS_F;
  float*  yy     = ws + WS_YY_F;
  __bf16* CH     = (__bf16*)((char*)d_ws + WS_CH_B);
  __bf16* CL     = (__bf16*)((char*)d_ws + WS_CL_B);
  double* accum  = (double*)((char*)d_ws + WS_ACC_B);

  nca_zero_kernel<<<(WS_ZERO_F + 255) / 256, 256, 0, stream>>>(ws, WS_ZERO_F, accum);
  nca_scatter_kernel<<<(n * (DIMS / 4) + 255) / 256, 256, 0, stream>>>(X, T, csum, counts, n);
  nca_finalize_centers<<<NUM_CLASS, 64, 0, stream>>>(csum, counts, yy, CH, CL);
  nca_main_kernel<<<n / (16 * 8), 256, 0, stream>>>(X, T, CH, CL, yy, accum);
  nca_writeout<<<1, 1, 0, stream>>>(accum, (float*)d_out, n);
}
